// AtomGNN_57483842290055
// MI455X (gfx1250) — compile-verified, matrix operand order checked by reading, not run on hardware
//
#include <hip/hip_runtime.h>

// AtomGNN on MI455X (gfx1250, wave32, WMMA f32_16x16x32_f16).
// Memory-bound design: edge MLP (80 GFLOP) runs on WMMA so the kernel sits at
// the HBM roofline (~2.4 GB of gather/scatter traffic -> ~105us at 23.3 TB/s).
// h kept in f16 (halves gather bytes), accumulate in f32, segment_sum fused as
// f32 global atomics straight from the WMMA C-fragments.

#define GN 50000
#define GE 1600000
#define WAVES 4
#define TPB (WAVES * 32)

typedef __attribute__((ext_vector_type(16))) _Float16 v16h;
typedef __attribute__((ext_vector_type(8)))  _Float16 v8h;
typedef __attribute__((ext_vector_type(8)))  float    v8f;

// Load a 16-bit A/B fragment from a row-major tile (ld = row stride in elems).
// Per ISA 16-bit A 16x32 layout: lane L -> row r=(L&15), k-base kb=(L>>4)*8;
// frag elems 0..7 = K kb..kb+7, elems 8..15 = K kb+16..kb+23.
// Caller passes r and kb (kb already includes the K-step offset).
__device__ __forceinline__ v16h ld_frag(const _Float16* t, int ld, int r, int kb) {
  union { v16h v; v8h h2[2]; } u;
  const _Float16* p = t + r * ld + kb;
  u.h2[0] = *(const v8h*)(p);
  u.h2[1] = *(const v8h*)(p + 16);
  return u.v;
}

__device__ __forceinline__ v8f wmma16(v16h a, v16h b, v8f c) {
  return __builtin_amdgcn_wmma_f32_16x16x32_f16(false, a, false, b, (short)0, c,
                                                false, false);
}

// ---------------- weight pre-pack: f32 [K,N] -> f16 [N,Kp] (B-fragment layout,
// i.e. transposed, K zero-padded to Kp) ----------------
__global__ void pack_wt(const float* __restrict__ src, _Float16* __restrict__ dst,
                        int K, int NN, int Kp) {
  int i = blockIdx.x * blockDim.x + threadIdx.x;
  if (i >= NN * Kp) return;
  int n = i / Kp, k = i % Kp;
  dst[i] = (k < K) ? (_Float16)src[k * NN + n] : (_Float16)0.f;
}

__global__ void zero_f4(float4* __restrict__ p, int n4) {
  int i = blockIdx.x * blockDim.x + threadIdx.x;
  if (i < n4) p[i] = make_float4(0.f, 0.f, 0.f, 0.f);
}

// ---------------- encoder: h = relu(x@W1+b1)@W2+b2, x:[N,16] ----------------
__global__ void __launch_bounds__(TPB) encoder_kernel(
    const float* __restrict__ x, const _Float16* __restrict__ w1t,
    const float* __restrict__ b1, const _Float16* __restrict__ w2t,
    const float* __restrict__ b2, float* __restrict__ h,
    _Float16* __restrict__ hb, int nTiles) {
  __shared__ _Float16 sW1[64 * 32];
  __shared__ _Float16 sW2[64 * 64];
  __shared__ _Float16 sA2[WAVES][16 * 64];
  const int tid = threadIdx.x;
  for (int i = tid; i < 64 * 32 * 2 / 16; i += TPB) ((uint4*)sW1)[i] = ((const uint4*)w1t)[i];
  for (int i = tid; i < 64 * 64 * 2 / 16; i += TPB) ((uint4*)sW2)[i] = ((const uint4*)w2t)[i];
  __syncthreads();
  const int wave = tid >> 5, lane = tid & 31;
  const int half = lane >> 4, l16 = lane & 15;
  _Float16* A2 = sA2[wave];
  float bb1[4], bb2[4];
#pragma unroll
  for (int nt = 0; nt < 4; ++nt) { bb1[nt] = b1[nt * 16 + l16]; bb2[nt] = b2[nt * 16 + l16]; }
  const int gw = blockIdx.x * WAVES + wave, gws = gridDim.x * WAVES;
  for (int t = gw; t < nTiles; t += gws) {
    const int base = t * 16;
    // Layer 1: K=16 padded to 32; frag hi-half (K 16..31) is zero padding.
    v16h a;
    const float* xr = x + (base + l16) * 16 + half * 8;
#pragma unroll
    for (int c = 0; c < 8; ++c) { a[c] = (_Float16)xr[c]; a[8 + c] = (_Float16)0.f; }
    v8f acc[4] = {};
#pragma unroll
    for (int nt = 0; nt < 4; ++nt) {
      v16h b = ld_frag(sW1, 32, nt * 16 + l16, half * 8);
      acc[nt] = wmma16(a, b, acc[nt]);
    }
#pragma unroll
    for (int nt = 0; nt < 4; ++nt)
#pragma unroll
      for (int j = 0; j < 8; ++j) {
        float v = acc[nt][j] + bb1[nt];
        v = v > 0.f ? v : 0.f;
        A2[(half * 8 + j) * 64 + nt * 16 + l16] = (_Float16)v;
      }
    // Layer 2: K=64 -> 2 K-steps.
    v8f acc2[4] = {};
#pragma unroll
    for (int kk = 0; kk < 2; ++kk) {
      v16h a2 = ld_frag(A2, 64, l16, kk * 32 + half * 8);
#pragma unroll
      for (int nt = 0; nt < 4; ++nt) {
        v16h b = ld_frag(sW2, 64, nt * 16 + l16, kk * 32 + half * 8);
        acc2[nt] = wmma16(a2, b, acc2[nt]);
      }
    }
#pragma unroll
    for (int nt = 0; nt < 4; ++nt)
#pragma unroll
      for (int j = 0; j < 8; ++j) {
        int row = base + half * 8 + j, col = nt * 16 + l16;
        float v = acc2[nt][j] + bb2[nt];
        h[row * 64 + col] = v;
        hb[row * 64 + col] = (_Float16)v;
      }
  }
}

// ---------------- edge MLP + fused segment_sum ----------------
// Per wave: tile of 16 edges. msg_in = [h[src] | h[dst] | ef] : 16x131 padded
// to 16x160 (5 K-steps of 32). Output 16x64 atomically scattered to agg[dst].
__global__ void __launch_bounds__(TPB) edge_kernel(
    const int* __restrict__ edges, const float* __restrict__ ef,
    const _Float16* __restrict__ hb, const _Float16* __restrict__ w1t,
    const float* __restrict__ b1, const _Float16* __restrict__ w2t,
    const float* __restrict__ b2, float* __restrict__ agg, int nTiles) {
  __shared__ _Float16 sW1[64 * 160];   // 20 KB
  __shared__ _Float16 sW2[64 * 64];    // 8 KB
  __shared__ _Float16 sA1[WAVES][16 * 160];
  __shared__ _Float16 sA2[WAVES][16 * 64];
  __shared__ int sDst[WAVES][16];
  const int tid = threadIdx.x;
  for (int i = tid; i < 64 * 160 * 2 / 16; i += TPB) ((uint4*)sW1)[i] = ((const uint4*)w1t)[i];
  for (int i = tid; i < 64 * 64 * 2 / 16; i += TPB) ((uint4*)sW2)[i] = ((const uint4*)w2t)[i];
  __syncthreads();
  const int wave = tid >> 5, lane = tid & 31;
  const int half = lane >> 4, l16 = lane & 15;
  _Float16* A1 = sA1[wave];
  _Float16* A2 = sA2[wave];
  // Zero K-padding cols [128,160) once (W1 rows >=131 are zero too; ef cols
  // 128..130 are rewritten every tile).
  for (int i = lane; i < 16 * 32; i += 32) {
    int r = i >> 5, c = i & 31;
    A1[r * 160 + 128 + c] = (_Float16)0.f;
  }
  float bb1[4], bb2[4];
#pragma unroll
  for (int nt = 0; nt < 4; ++nt) { bb1[nt] = b1[nt * 16 + l16]; bb2[nt] = b2[nt * 16 + l16]; }
  const int gw = blockIdx.x * WAVES + wave, gws = gridDim.x * WAVES;
  for (int t = gw; t < nTiles; t += gws) {
    const int base = t * 16;
    const int e = base + l16;
    const int2 pr = *(const int2*)(edges + 2 * e);  // (src, dst)
    const int node = half ? pr.y : pr.x;
    {  // lane gathers one 128-byte f16 row: lanes 0-15 -> h[src], 16-31 -> h[dst]
      const _Float16* srow = hb + (size_t)node * 64;
      _Float16* drow = A1 + l16 * 160 + half * 64;
#pragma unroll
      for (int i = 0; i < 8; ++i) *(v8h*)(drow + i * 8) = *(const v8h*)(srow + i * 8);
    }
    if (!half) {
      sDst[wave][l16] = pr.y;
      const float* e3 = ef + (size_t)e * 3;
      A1[l16 * 160 + 128] = (_Float16)e3[0];
      A1[l16 * 160 + 129] = (_Float16)e3[1];
      A1[l16 * 160 + 130] = (_Float16)e3[2];
    }
    // Layer 1: 5 K-steps x 4 N-tiles = 20 WMMAs.
    v8f acc[4] = {};
#pragma unroll
    for (int kk = 0; kk < 5; ++kk) {
      v16h a = ld_frag(A1, 160, l16, kk * 32 + half * 8);
#pragma unroll
      for (int nt = 0; nt < 4; ++nt) {
        v16h b = ld_frag(sW1, 160, nt * 16 + l16, kk * 32 + half * 8);
        acc[nt] = wmma16(a, b, acc[nt]);
      }
    }
#pragma unroll
    for (int nt = 0; nt < 4; ++nt)
#pragma unroll
      for (int j = 0; j < 8; ++j) {
        float v = acc[nt][j] + bb1[nt];
        v = v > 0.f ? v : 0.f;
        A2[(half * 8 + j) * 64 + nt * 16 + l16] = (_Float16)v;
      }
    // Layer 2: 2 K-steps x 4 N-tiles = 8 WMMAs.
    v8f acc2[4] = {};
#pragma unroll
    for (int kk = 0; kk < 2; ++kk) {
      v16h a2 = ld_frag(A2, 64, l16, kk * 32 + half * 8);
#pragma unroll
      for (int nt = 0; nt < 4; ++nt) {
        v16h b = ld_frag(sW2, 64, nt * 16 + l16, kk * 32 + half * 8);
        acc2[nt] = wmma16(a2, b, acc2[nt]);
      }
    }
    int didx[8];
#pragma unroll
    for (int j = 0; j < 8; ++j) didx[j] = sDst[wave][half * 8 + j];
#pragma unroll
    for (int nt = 0; nt < 4; ++nt)
#pragma unroll
      for (int j = 0; j < 8; ++j) {
        float v = acc2[nt][j] + bb2[nt];
        atomicAdd(&agg[(size_t)didx[j] * 64 + nt * 16 + l16], v);
      }
  }
}

// ---------------- update: h += mlp2([h | agg]) ----------------
__device__ __forceinline__ v8h ld_chunk(const _Float16* hb, const float* agg,
                                        int node, int c) {
  if (c < 64) return *(const v8h*)(hb + (size_t)node * 64 + c);
  v8h r;
  const float* a = agg + (size_t)node * 64 + (c - 64);
#pragma unroll
  for (int i = 0; i < 8; ++i) r[i] = (_Float16)a[i];
  return r;
}

__global__ void __launch_bounds__(TPB) update_kernel(
    float* __restrict__ h, _Float16* __restrict__ hb,
    const float* __restrict__ agg, const _Float16* __restrict__ w1t,
    const float* __restrict__ b1, const _Float16* __restrict__ w2t,
    const float* __restrict__ b2, int nTiles) {
  __shared__ _Float16 sW1[64 * 128];
  __shared__ _Float16 sW2[64 * 64];
  __shared__ _Float16 sA2[WAVES][16 * 64];
  const int tid = threadIdx.x;
  for (int i = tid; i < 64 * 128 * 2 / 16; i += TPB) ((uint4*)sW1)[i] = ((const uint4*)w1t)[i];
  for (int i = tid; i < 64 * 64 * 2 / 16; i += TPB) ((uint4*)sW2)[i] = ((const uint4*)w2t)[i];
  __syncthreads();
  const int wave = tid >> 5, lane = tid & 31;
  const int half = lane >> 4, l16 = lane & 15;
  _Float16* A2 = sA2[wave];
  float bb1[4], bb2[4];
#pragma unroll
  for (int nt = 0; nt < 4; ++nt) { bb1[nt] = b1[nt * 16 + l16]; bb2[nt] = b2[nt * 16 + l16]; }
  const int gw = blockIdx.x * WAVES + wave, gws = gridDim.x * WAVES;
  for (int t = gw; t < nTiles; t += gws) {
    const int base = t * 16;
    const int node = base + l16;  // this lane's A-row
    // Layer 1: K=128 -> 4 K-steps; A-frags assembled in registers from hb|agg.
    v8f acc[4] = {};
#pragma unroll
    for (int kk = 0; kk < 4; ++kk) {
      const int c0 = kk * 32 + half * 8;
      union { v16h v; v8h h2[2]; } u;
      u.h2[0] = ld_chunk(hb, agg, node, c0);
      u.h2[1] = ld_chunk(hb, agg, node, c0 + 16);
#pragma unroll
      for (int nt = 0; nt < 4; ++nt) {
        v16h b = ld_frag(sW1, 128, nt * 16 + l16, c0);
        acc[nt] = wmma16(u.v, b, acc[nt]);
      }
    }
#pragma unroll
    for (int nt = 0; nt < 4; ++nt)
#pragma unroll
      for (int j = 0; j < 8; ++j) {
        float v = acc[nt][j] + bb1[nt];
        v = v > 0.f ? v : 0.f;
        A2[(half * 8 + j) * 64 + nt * 16 + l16] = (_Float16)v;
      }
    v8f acc2[4] = {};
#pragma unroll
    for (int kk = 0; kk < 2; ++kk) {
      v16h a2 = ld_frag(A2, 64, l16, kk * 32 + half * 8);
#pragma unroll
      for (int nt = 0; nt < 4; ++nt) {
        v16h b = ld_frag(sW2, 64, nt * 16 + l16, kk * 32 + half * 8);
        acc2[nt] = wmma16(a2, b, acc2[nt]);
      }
    }
#pragma unroll
    for (int nt = 0; nt < 4; ++nt)
#pragma unroll
      for (int j = 0; j < 8; ++j) {
        int row = base + half * 8 + j, col = nt * 16 + l16;
        float v = h[row * 64 + col] + acc2[nt][j] + bb2[nt];
        h[row * 64 + col] = v;
        hb[row * 64 + col] = (_Float16)v;
      }
  }
}

// ---------------- head: out = (relu(h@W1+b1)@w2 + b2)[:,0] ----------------
__global__ void __launch_bounds__(TPB) head_kernel(
    const _Float16* __restrict__ hb, const _Float16* __restrict__ w1t,
    const float* __restrict__ b1, const float* __restrict__ w2,
    const float* __restrict__ b2, float* __restrict__ out, int nTiles) {
  __shared__ _Float16 sW1[64 * 64];
  __shared__ float sw2[64];
  __shared__ float sT[WAVES][16 * 64];
  const int tid = threadIdx.x;
  for (int i = tid; i < 64 * 64 * 2 / 16; i += TPB) ((uint4*)sW1)[i] = ((const uint4*)w1t)[i];
  if (tid < 16) ((uint4*)sw2)[tid] = ((const uint4*)w2)[tid];
  __syncthreads();
  const int wave = tid >> 5, lane = tid & 31;
  const int half = lane >> 4, l16 = lane & 15;
  float* T = sT[wave];
  float bb1[4];
#pragma unroll
  for (int nt = 0; nt < 4; ++nt) bb1[nt] = b1[nt * 16 + l16];
  const float b2v = b2[0];
  const int gw = blockIdx.x * WAVES + wave, gws = gridDim.x * WAVES;
  for (int t = gw; t < nTiles; t += gws) {
    const int base = t * 16;
    v8f acc[4] = {};
#pragma unroll
    for (int kk = 0; kk < 2; ++kk) {
      v16h a = ld_frag(hb + (size_t)base * 64, 64, l16, kk * 32 + half * 8);
#pragma unroll
      for (int nt = 0; nt < 4; ++nt) {
        v16h b = ld_frag(sW1, 64, nt * 16 + l16, kk * 32 + half * 8);
        acc[nt] = wmma16(a, b, acc[nt]);
      }
    }
#pragma unroll
    for (int nt = 0; nt < 4; ++nt)
#pragma unroll
      for (int j = 0; j < 8; ++j) {
        float v = acc[nt][j] + bb1[nt];
        T[(half * 8 + j) * 64 + nt * 16 + l16] = v > 0.f ? v : 0.f;
      }
    // out[m] = b2 + sum_n T[m][n]*w2[n]; each lane sums half a row, pair-reduce.
    float s = 0.f;
#pragma unroll
    for (int c = 0; c < 32; ++c) s += T[l16 * 64 + half * 32 + c] * sw2[half * 32 + c];
    s += __shfl_xor(s, 16, 32);
    if (!half) out[base + l16] = s + b2v;
  }
}

extern "C" void kernel_launch(void* const* d_in, const int* in_sizes, int n_in,
                              void* d_out, int out_size, void* d_ws, size_t ws_size,
                              hipStream_t stream) {
  (void)in_sizes; (void)n_in; (void)out_size; (void)ws_size;
  const float* x       = (const float*)d_in[0];
  const int*   edges   = (const int*)  d_in[1];
  const float* ef      = (const float*)d_in[2];
  const float* enc_w1  = (const float*)d_in[3];
  const float* enc_b1  = (const float*)d_in[4];
  const float* enc_w2  = (const float*)d_in[5];
  const float* enc_b2  = (const float*)d_in[6];
  const float* msg_w1  = (const float*)d_in[7];
  const float* msg_b1  = (const float*)d_in[8];
  const float* msg_w2  = (const float*)d_in[9];
  const float* msg_b2  = (const float*)d_in[10];
  const float* upd_w1  = (const float*)d_in[11];
  const float* upd_b1  = (const float*)d_in[12];
  const float* upd_w2  = (const float*)d_in[13];
  const float* upd_b2  = (const float*)d_in[14];
  const float* head_w1 = (const float*)d_in[15];
  const float* head_b1 = (const float*)d_in[16];
  const float* head_w2 = (const float*)d_in[17];
  const float* head_b2 = (const float*)d_in[18];

  // Workspace carve (~32.2 MB), all chunks 16B-aligned.
  char* w = (char*)d_ws;
  float*    h   = (float*)w;    w += (size_t)GN * 64 * 4;
  _Float16* hb  = (_Float16*)w; w += (size_t)GN * 64 * 2;
  float*    agg = (float*)w;    w += (size_t)GN * 64 * 4;
  _Float16* enc_w1t  = (_Float16*)w; w += 64 * 32 * 2;
  _Float16* enc_w2t  = (_Float16*)w; w += 64 * 64 * 2;
  _Float16* msg_w1t  = (_Float16*)w; w += 2 * 64 * 160 * 2;
  _Float16* msg_w2t  = (_Float16*)w; w += 2 * 64 * 64 * 2;
  _Float16* upd_w1t  = (_Float16*)w; w += 2 * 64 * 128 * 2;
  _Float16* upd_w2t  = (_Float16*)w; w += 2 * 64 * 64 * 2;
  _Float16* head_w1t = (_Float16*)w; w += 64 * 64 * 2;

  auto pk = [&](const float* s, _Float16* d, int K, int NN, int Kp) {
    int tot = NN * Kp;
    pack_wt<<<(tot + 255) / 256, 256, 0, stream>>>(s, d, K, NN, Kp);
  };
  pk(enc_w1, enc_w1t, 16, 64, 32);
  pk(enc_w2, enc_w2t, 64, 64, 64);
  for (int r = 0; r < 2; ++r) {
    pk(msg_w1 + r * 131 * 64, msg_w1t + r * 64 * 160, 131, 64, 160);
    pk(msg_w2 + r * 64 * 64,  msg_w2t + r * 64 * 64,  64, 64, 64);
    pk(upd_w1 + r * 128 * 64, upd_w1t + r * 64 * 128, 128, 64, 128);
    pk(upd_w2 + r * 64 * 64,  upd_w2t + r * 64 * 64,  64, 64, 64);
  }
  pk(head_w1, head_w1t, 64, 64, 64);

  encoder_kernel<<<512, TPB, 0, stream>>>(x, enc_w1t, enc_b1, enc_w2t, enc_b2,
                                          h, hb, GN / 16);
  for (int r = 0; r < 2; ++r) {
    zero_f4<<<((GN * 64 / 4) + 255) / 256, 256, 0, stream>>>((float4*)agg, GN * 64 / 4);
    edge_kernel<<<2048, TPB, 0, stream>>>(edges, ef, hb,
        msg_w1t + r * 64 * 160, msg_b1 + r * 64,
        msg_w2t + r * 64 * 64,  msg_b2 + r * 64, agg, GE / 16);
    update_kernel<<<512, TPB, 0, stream>>>(h, hb, agg,
        upd_w1t + r * 64 * 128, upd_b1 + r * 64,
        upd_w2t + r * 64 * 64,  upd_b2 + r * 64, GN / 16);
  }
  head_kernel<<<512, TPB, 0, stream>>>(hb, head_w1t, head_b1, head_w2, head_b2,
                                       (float*)d_out, GN / 16);
}
